// MultiHeadAttention_17626545783293
// MI455X (gfx1250) — compile-verified, matrix operand order checked by reading, not run on hardware
//
#include <hip/hip_runtime.h>
#include <hip/hip_bf16.h>

typedef __attribute__((ext_vector_type(16))) _Float16 v16h;
typedef __attribute__((ext_vector_type(8)))  _Float16 v8h;
typedef __attribute__((ext_vector_type(8)))  float    v8f;
typedef __attribute__((ext_vector_type(4)))  unsigned int u32x4;
typedef __attribute__((ext_vector_type(8)))  int i32x8;
typedef __attribute__((ext_vector_type(4)))  int i32x4;

#define D_MODEL   1024
#define NUM_HEADS 16
#define HEAD_DIM  64
#define BATCH     2
#define SEQ       2048
#define MTOT      (BATCH * SEQ)   /* 4096 rows of (b,s) */

#define BLK_M 128
#define BLK_N 64
#define BLK_K 32
#define LPAD  8                  /* +8 halves (16B) row pad: kills LDS bank conflicts */

#define FM    32                 /* fused kernel: score rows per block */
#define KCH   128                /* K rows staged per TDM chunk */
#define SP_LD (SEQ + LPAD)       /* 2056 halves */
#define SQ_LD (HEAD_DIM + LPAD)  /* 72 halves  */

// ---------------------------------------------------------------------------
// WMMA helpers (CDNA5 wave32, V_WMMA_F32_16X16X32_F16)
// ---------------------------------------------------------------------------
__device__ __forceinline__ v8f wmma16(v16h a, v16h b, v8f c) {
  return __builtin_amdgcn_wmma_f32_16x16x32_f16(false, a, false, b, (short)0, c,
                                                false, false);
}

// 16x32 f16 fragment from a row-major tile (leading dim `ld` halves).
// ISA 7.12.2: lane L holds row M=L%16; lanes 0-15 cover K {0..7,16..23},
// lanes 16-31 cover K {8..15,24..31}; VGPR v packs K = base+2v, 2v+1.
__device__ __forceinline__ v16h load_frag(const _Float16* p, int ld) {
  const int lane = threadIdx.x & 31;
  const _Float16* r = p + (lane & 15) * ld + ((lane & 16) ? 8 : 0);
  v16h f;
#pragma unroll
  for (int v = 0; v < 8; ++v) {
    const int k = ((v & 4) ? 16 : 0) + (v & 3) * 2;
    f[2 * v]     = r[k];
    f[2 * v + 1] = r[k + 1];
  }
  return f;
}

// ---------------------------------------------------------------------------
// TDM: async 2D tile load Global -> LDS (TENSOR_LOAD_TO_LDS).
// Tile = KCH x 64 f16 rows from a row-major [S][64] tensor; hardware padding
// (pad_interval=32 DWORDs = one 128B row, pad_amount=4 DWORDs = 16B) lands the
// tile directly in the bank-conflict-free SQ_LD=72-half row layout.
// D# fields per CDNA5 ISA ch.8 (group0 128b, group1 256b, groups2/3 unused 2D).
// ---------------------------------------------------------------------------
__device__ __forceinline__ void tdm_load_k(const _Float16* gsrc, void* ldst) {
  const unsigned lds_off = (unsigned)(size_t)ldst;  // LDS aperture: low 32 bits
  const unsigned long long ga = (unsigned long long)(size_t)gsrc;
  u32x4 g0;
  g0[0] = 1u;                                        // count=1 (valid user D#)
  g0[1] = lds_off;                                   // lds_addr (bytes)
  g0[2] = (unsigned)(ga & 0xffffffffu);              // global_addr[31:0]
  g0[3] = (unsigned)((ga >> 32) & 0x01ffffffu)       // global_addr[56:32]
          | (2u << 30);                              // type=2 ("image")
  i32x8 g1;
  g1[0] = (int)((1u << 16)      // data_size = 2 bytes
              | (1u << 20)      // pad_enable
              | (4u << 22)      // pad_interval: 32 DWORDs (= one 128B row)
              | (3u << 25));    // pad_amount: 4 DWORDs (= 16B)
  g1[1] = (int)((unsigned)HEAD_DIM << 16);           // tensor_dim0[15:0] @ bits79:48
  g1[2] = (int)((unsigned)SEQ << 16);                // tensor_dim1[15:0] @ bits111:80
  g1[3] = (int)((unsigned)HEAD_DIM << 16);           // tile_dim0 @ bits127:112
  g1[4] = (int)(unsigned)KCH;                        // tile_dim1 @ bits143:128
  g1[5] = (int)(unsigned)HEAD_DIM;                   // tensor_dim0_stride low32
  g1[6] = 0;
  g1[7] = 0;
  const i32x4 gz = {0, 0, 0, 0};
#if defined(__clang_major__) && (__clang_major__ >= 23)
  const i32x8 gz8 = {0, 0, 0, 0, 0, 0, 0, 0};
  __builtin_amdgcn_tensor_load_to_lds(g0, g1, gz, gz, gz8, 0);
#else
  __builtin_amdgcn_tensor_load_to_lds(g0, g1, gz, gz, 0);
#endif
}

// ---------------------------------------------------------------------------
// Kernel 1: convert W (f32, [N][K]) -> W^T (f16, [K][N])
// ---------------------------------------------------------------------------
__global__ __launch_bounds__(256) void mha_convert_wT(
    const float* __restrict__ W, _Float16* __restrict__ WT) {
  const int idx = blockIdx.x * 256 + threadIdx.x;
  const int kk = idx >> 10;
  const int n  = idx & 1023;
  WT[idx] = (_Float16)W[(size_t)n * D_MODEL + kk];
}

// ---------------------------------------------------------------------------
// Kernel 2: projection GEMM  Y = X @ W^T + b, scattered into head layout.
//   kt_mode == 0 : dst[b][h][s][d]   (Q, K)
//   kt_mode == 1 : dst[b][h][d][s]   (V transposed, B-operand for PV)
// ---------------------------------------------------------------------------
__global__ __launch_bounds__(256) void mha_proj_kernel(
    const float* __restrict__ X, const _Float16* __restrict__ WT,
    const float* __restrict__ bias, _Float16* __restrict__ dst, int kt_mode) {
  __shared__ __align__(16) _Float16 sA[BLK_M][BLK_K + LPAD];
  __shared__ __align__(16) _Float16 sB[BLK_N][BLK_K + LPAD];
  const int tid  = threadIdx.x;
  const int lane = tid & 31;
  const int w    = tid >> 5;
  const int m0 = blockIdx.y * BLK_M;
  const int n0 = blockIdx.x * BLK_N;
  const int am = (w >> 1) * 32;
  const int bn = (w & 1) * 32;

  const v8f vz = {0.f, 0.f, 0.f, 0.f, 0.f, 0.f, 0.f, 0.f};
  v8f acc[2][2];
#pragma unroll
  for (int i = 0; i < 2; ++i)
#pragma unroll
    for (int j = 0; j < 2; ++j) acc[i][j] = vz;

  for (int k0 = 0; k0 < D_MODEL; k0 += BLK_K) {
#pragma unroll
    for (int p = 0; p < 4; ++p) {
      const int m  = p * 32 + (tid >> 3);
      const int kc = (tid & 7) * 4;
      const float4 vv =
          *(const float4*)(X + (size_t)(m0 + m) * D_MODEL + k0 + kc);
      sA[m][kc + 0] = (_Float16)vv.x;
      sA[m][kc + 1] = (_Float16)vv.y;
      sA[m][kc + 2] = (_Float16)vv.z;
      sA[m][kc + 3] = (_Float16)vv.w;
    }
    {
      const int kk = tid >> 3;
      const int nc = (tid & 7) * 8;
      const v8h vv =
          *(const v8h*)(WT + (size_t)(k0 + kk) * D_MODEL + n0 + nc);
#pragma unroll
      for (int i = 0; i < 8; ++i) sB[nc + i][kk] = vv[i];
    }
    __syncthreads();
    const v16h a0 = load_frag(&sA[am][0], BLK_K + LPAD);
    const v16h a1 = load_frag(&sA[am + 16][0], BLK_K + LPAD);
    const v16h b0 = load_frag(&sB[bn][0], BLK_K + LPAD);
    const v16h b1 = load_frag(&sB[bn + 16][0], BLK_K + LPAD);
    acc[0][0] = wmma16(a0, b0, acc[0][0]);
    acc[0][1] = wmma16(a0, b1, acc[0][1]);
    acc[1][0] = wmma16(a1, b0, acc[1][0]);
    acc[1][1] = wmma16(a1, b1, acc[1][1]);
    __syncthreads();
  }

  const int mrow = (lane & 16) ? 8 : 0;
  const int ncol = lane & 15;
#pragma unroll
  for (int i = 0; i < 2; ++i)
#pragma unroll
    for (int j = 0; j < 2; ++j)
#pragma unroll
      for (int r = 0; r < 8; ++r) {
        const int m = m0 + am + i * 16 + mrow + r;
        const int n = n0 + bn + j * 16 + ncol;
        const float val = acc[i][j][r] + bias[n];
        const int bb = m >> 11, ss = m & (SEQ - 1);
        const int hh = n >> 6, dd = n & (HEAD_DIM - 1);
        const size_t off =
            kt_mode
                ? ((size_t)((bb * NUM_HEADS + hh) * HEAD_DIM + dd)) * SEQ + ss
                : ((size_t)((bb * NUM_HEADS + hh) * SEQ + ss)) * HEAD_DIM + dd;
        dst[off] = (_Float16)val;
      }
}

// ---------------------------------------------------------------------------
// Kernel 3 (fused attention): per (b,h) and 32-row strip:
//   TDM double-buffered K staging -> scores (WMMA) -> LDS f16 strip ->
//   softmax stats -> single normalized attn write -> PV (WMMA) -> AO f16.
// LDS: sP 128.5KB + sK 36KB + sQ 4.5KB + scratch (~170KB of 320KB WGP).
// ---------------------------------------------------------------------------
__global__ __launch_bounds__(256) void mha_fused_attn_kernel(
    const _Float16* __restrict__ Qh,   // [B,H,S,hd]
    const _Float16* __restrict__ Kh,   // [B,H,S,hd]
    const _Float16* __restrict__ VhT,  // [B,H,hd,S]
    float* __restrict__ attn,          // [B,H,S,S]
    _Float16* __restrict__ AO) {       // [B,S,D]
  __shared__ __align__(16) _Float16 sP[FM][SP_LD];
  __shared__ __align__(16) _Float16 sQ[FM][SQ_LD];
  __shared__ __align__(16) _Float16 sK[2][KCH][SQ_LD];
  __shared__ float sRed[FM][8];
  __shared__ float sMax[FM];
  __shared__ float sInv[FM];

  const int tid  = threadIdx.x;
  const int lane = tid & 31;
  const int w    = tid >> 5;
  const int bh   = blockIdx.z;
  const int bb   = bh >> 4, hh = bh & 15;
  const int m0   = blockIdx.x * FM;

  const _Float16* Qb = Qh + (size_t)bh * SEQ * HEAD_DIM;
  const _Float16* Kb = Kh + (size_t)bh * SEQ * HEAD_DIM;
  const _Float16* Vb = VhT + (size_t)bh * HEAD_DIM * SEQ;

  const int mrow = (lane & 16) ? 8 : 0;
  const int ncol = lane & 15;
  const v8f vz = {0.f, 0.f, 0.f, 0.f, 0.f, 0.f, 0.f, 0.f};

  // ---- stage Q strip: 32 x 64 f16 ----
  {
    const int m  = tid >> 3;
    const int kc = (tid & 7) * 8;
    *(v8h*)&sQ[m][kc] =
        *(const v8h*)(Qb + (size_t)(m0 + m) * HEAD_DIM + kc);
  }

  // ---- phase 1: scores = (Q K^T)/8 into sP, K staged by TDM double buffer --
  {
    const int wm = (w >> 2) * 16;   // 0 / 16
    const int wn = (w & 3) * 32;    // 0..96 within chunk
    const int NCH = SEQ / KCH;      // 16 chunks
    if (w == 0) tdm_load_k(Kb, &sK[0][0][0]);
    __syncthreads();                // sQ visible too
    for (int c = 0; c < NCH; ++c) {
      const int cb = c & 1;
      if (w == 0) __builtin_amdgcn_s_wait_tensorcnt(0);
      __syncthreads();              // chunk c resident for all waves
      if (w == 0 && (c + 1) < NCH)
        tdm_load_k(Kb + (size_t)(c + 1) * KCH * HEAD_DIM, &sK[cb ^ 1][0][0]);
      v8f acc0 = vz, acc1 = vz;
#pragma unroll
      for (int k0 = 0; k0 < HEAD_DIM; k0 += BLK_K) {
        const v16h a  = load_frag(&sQ[wm][k0], SQ_LD);
        const v16h b0 = load_frag(&sK[cb][wn][k0], SQ_LD);
        const v16h b1 = load_frag(&sK[cb][wn + 16][k0], SQ_LD);
        acc0 = wmma16(a, b0, acc0);
        acc1 = wmma16(a, b1, acc1);
      }
      const int n0 = c * KCH + wn;
#pragma unroll
      for (int r = 0; r < 8; ++r) {
        const int rl = wm + mrow + r;
        sP[rl][n0 + ncol]      = (_Float16)(acc0[r] * 0.125f);
        sP[rl][n0 + 16 + ncol] = (_Float16)(acc1[r] * 0.125f);
      }
      __syncthreads();              // chunk c consumed before buffer reuse
    }
  }

  // ---- phase 2: softmax stats; sP <- exp(s - max) (unnormalized, f16) ----
  {
    const int row = tid >> 3;    // 0..31
    const int seg = tid & 7;     // 0..7, each 256 cols
    float mx = -3.0e38f;
#pragma unroll 4
    for (int i = 0; i < 32; ++i) {
      const v8h v = *(const v8h*)&sP[row][seg * 256 + i * 8];
#pragma unroll
      for (int e = 0; e < 8; ++e) mx = fmaxf(mx, (float)v[e]);
    }
    sRed[row][seg] = mx;
    __syncthreads();
    if (tid < FM) {
      float m2 = sRed[tid][0];
#pragma unroll
      for (int e = 1; e < 8; ++e) m2 = fmaxf(m2, sRed[tid][e]);
      sMax[tid] = m2;
    }
    __syncthreads();
    const float rmax = sMax[row];
    float sum = 0.f;
#pragma unroll 4
    for (int i = 0; i < 32; ++i) {
      v8h v = *(v8h*)&sP[row][seg * 256 + i * 8];
      v8h o;
#pragma unroll
      for (int e = 0; e < 8; ++e) {
        const float ev = __expf((float)v[e] - rmax);
        sum += ev;
        o[e] = (_Float16)ev;
      }
      *(v8h*)&sP[row][seg * 256 + i * 8] = o;
    }
    sRed[row][seg] = sum;
    __syncthreads();
    if (tid < FM) {
      float s2 = 0.f;
#pragma unroll
      for (int e = 0; e < 8; ++e) s2 += sRed[tid][e];
      sInv[tid] = 1.0f / s2;
    }
    __syncthreads();
  }

  // ---- phase 3: single normalized attn write (f32, coalesced) ----
  {
    float* attnp = attn + (size_t)bh * SEQ * SEQ + (size_t)m0 * SEQ;
    const int col = tid * 8;
    for (int r = 0; r < FM; ++r) {
      const v8h v = *(const v8h*)&sP[r][col];
      const float inv = sInv[r];
      float4 o0, o1;
      o0.x = (float)v[0] * inv; o0.y = (float)v[1] * inv;
      o0.z = (float)v[2] * inv; o0.w = (float)v[3] * inv;
      o1.x = (float)v[4] * inv; o1.y = (float)v[5] * inv;
      o1.z = (float)v[6] * inv; o1.w = (float)v[7] * inv;
      *(float4*)(attnp + (size_t)r * SEQ + col)     = o0;
      *(float4*)(attnp + (size_t)r * SEQ + col + 4) = o1;
    }
  }

  // ---- phase 4: PV — A from LDS strip, B = V^T from global; scale by 1/l ----
  {
    const int wi = (w >> 2) * 16;   // row tile (0/16)
    const int dn = (w & 3) * 16;    // head-dim tile (0..48)
    const _Float16* VT = Vb + (size_t)dn * SEQ;
    v8f acc0 = vz, acc1 = vz;
    for (int k0 = 0; k0 < SEQ; k0 += 64) {
      if (k0 + 64 < SEQ) __builtin_prefetch(VT + k0 + 64, 0, 0);
      const v16h a0 = load_frag(&sP[wi][k0], SP_LD);
      const v16h b0 = load_frag(VT + k0, SEQ);
      acc0 = wmma16(a0, b0, acc0);
      const v16h a1 = load_frag(&sP[wi][k0 + 32], SP_LD);
      const v16h b1 = load_frag(VT + k0 + 32, SEQ);
      acc1 = wmma16(a1, b1, acc1);
    }
#pragma unroll
    for (int r = 0; r < 8; ++r) {
      const int rl = wi + mrow + r;
      const int m  = m0 + rl;
      const float val = (acc0[r] + acc1[r]) * sInv[rl];
      AO[((size_t)(bb * SEQ + m)) * D_MODEL + hh * HEAD_DIM + dn + ncol] =
          (_Float16)val;
    }
  }
}

// ---------------------------------------------------------------------------
// Kernel 4: Out = AO @ Wo^T + bo  (f32 to d_out)
// ---------------------------------------------------------------------------
__global__ __launch_bounds__(256) void mha_final_kernel(
    const _Float16* __restrict__ AO, const _Float16* __restrict__ WoT,
    const float* __restrict__ bo, float* __restrict__ outp) {
  __shared__ __align__(16) _Float16 sA[BLK_M][BLK_K + LPAD];
  __shared__ __align__(16) _Float16 sB[BLK_N][BLK_K + LPAD];
  const int tid = threadIdx.x, lane = tid & 31, w = tid >> 5;
  const int m0 = blockIdx.y * BLK_M;
  const int n0 = blockIdx.x * BLK_N;
  const int am = (w >> 1) * 32, bn = (w & 1) * 32;

  const v8f vz = {0.f, 0.f, 0.f, 0.f, 0.f, 0.f, 0.f, 0.f};
  v8f acc[2][2];
#pragma unroll
  for (int i = 0; i < 2; ++i)
#pragma unroll
    for (int j = 0; j < 2; ++j) acc[i][j] = vz;

  for (int k0 = 0; k0 < D_MODEL; k0 += BLK_K) {
#pragma unroll
    for (int p = 0; p < 2; ++p) {
      const int m  = p * 64 + (tid >> 2);
      const int kc = (tid & 3) * 8;
      *(v8h*)&sA[m][kc] =
          *(const v8h*)(AO + (size_t)(m0 + m) * D_MODEL + k0 + kc);
    }
    {
      const int kk = tid >> 3;
      const int nc = (tid & 7) * 8;
      const v8h vv =
          *(const v8h*)(WoT + (size_t)(k0 + kk) * D_MODEL + n0 + nc);
#pragma unroll
      for (int i = 0; i < 8; ++i) sB[nc + i][kk] = vv[i];
    }
    __syncthreads();
    const v16h a0 = load_frag(&sA[am][0], BLK_K + LPAD);
    const v16h a1 = load_frag(&sA[am + 16][0], BLK_K + LPAD);
    const v16h b0 = load_frag(&sB[bn][0], BLK_K + LPAD);
    const v16h b1 = load_frag(&sB[bn + 16][0], BLK_K + LPAD);
    acc[0][0] = wmma16(a0, b0, acc[0][0]);
    acc[0][1] = wmma16(a0, b1, acc[0][1]);
    acc[1][0] = wmma16(a1, b0, acc[1][0]);
    acc[1][1] = wmma16(a1, b1, acc[1][1]);
    __syncthreads();
  }

  const int mrow = (lane & 16) ? 8 : 0;
  const int ncol = lane & 15;
#pragma unroll
  for (int i = 0; i < 2; ++i)
#pragma unroll
    for (int j = 0; j < 2; ++j)
#pragma unroll
      for (int r = 0; r < 8; ++r) {
        const int m = m0 + am + i * 16 + mrow + r;
        const int n = n0 + bn + j * 16 + ncol;
        outp[(size_t)m * D_MODEL + n] = acc[i][j][r] + bo[n];
      }
}

// ---------------------------------------------------------------------------
// Host launcher
// ---------------------------------------------------------------------------
extern "C" void kernel_launch(void* const* d_in, const int* in_sizes, int n_in,
                              void* d_out, int out_size, void* d_ws,
                              size_t ws_size, hipStream_t stream) {
  const float* q  = (const float*)d_in[0];
  const float* k  = (const float*)d_in[1];
  const float* v  = (const float*)d_in[2];
  const float* Wq = (const float*)d_in[3];
  const float* bq = (const float*)d_in[4];
  const float* Wk = (const float*)d_in[5];
  const float* bk = (const float*)d_in[6];
  const float* Wv = (const float*)d_in[7];
  const float* bv = (const float*)d_in[8];
  const float* Wo = (const float*)d_in[9];
  const float* bo = (const float*)d_in[10];

  float* out  = (float*)d_out;
  float* attn = out + (size_t)BATCH * SEQ * D_MODEL;

  _Float16* ws = (_Float16*)d_ws;
  const size_t WSZ = (size_t)D_MODEL * D_MODEL;       // 1M halves
  const size_t HSZ = (size_t)BATCH * SEQ * D_MODEL;   // 4M halves
  _Float16* WqT = ws;
  _Float16* WkT = WqT + WSZ;
  _Float16* WvT = WkT + WSZ;
  _Float16* WoT = WvT + WSZ;
  _Float16* Qh  = WoT + WSZ;   // [B,H,S,hd]
  _Float16* Kh  = Qh + HSZ;    // [B,H,S,hd]
  _Float16* VhT = Kh + HSZ;    // [B,H,hd,S]
  _Float16* AO  = VhT + HSZ;   // [B,S,D]

  // 1) weights -> f16 transposed
  const dim3 cgrid((unsigned)(WSZ / 256));
  mha_convert_wT<<<cgrid, 256, 0, stream>>>(Wq, WqT);
  mha_convert_wT<<<cgrid, 256, 0, stream>>>(Wk, WkT);
  mha_convert_wT<<<cgrid, 256, 0, stream>>>(Wv, WvT);
  mha_convert_wT<<<cgrid, 256, 0, stream>>>(Wo, WoT);

  // 2) projections (Q,K row-major head layout; V transposed)
  const dim3 pgrid(D_MODEL / BLK_N, MTOT / BLK_M);
  mha_proj_kernel<<<pgrid, 256, 0, stream>>>(q, WqT, bq, Qh, 0);
  mha_proj_kernel<<<pgrid, 256, 0, stream>>>(k, WkT, bk, Kh, 0);
  mha_proj_kernel<<<pgrid, 256, 0, stream>>>(v, WvT, bv, VhT, 1);

  // 3) fused scores + softmax + attn write + PV (TDM-staged K)
  const dim3 agrid(SEQ / FM, 1, BATCH * NUM_HEADS);
  mha_fused_attn_kernel<<<agrid, 256, 0, stream>>>(Qh, Kh, VhT, attn, AO);

  // 4) output projection
  const dim3 fgrid(D_MODEL / BLK_N, MTOT / BLK_M);
  mha_final_kernel<<<fgrid, 256, 0, stream>>>(AO, WoT, bo, out);
}
